// TrajectoryLSTM_25907242729736
// MI455X (gfx1250) — matrix operand order repaired and checked by reading, not compiled, not run on hardware
//
#include <hip/hip_runtime.h>

typedef __attribute__((ext_vector_type(16))) _Float16 v16h;
typedef __attribute__((ext_vector_type(8)))  float    v8f;
typedef __attribute__((ext_vector_type(4)))  unsigned int v4u;
typedef __attribute__((ext_vector_type(8)))  int      v8i;
typedef __attribute__((ext_vector_type(4)))  int      v4i;

#define B_TOT    4096
#define T_ENC    200
#define T_DEC    200
#define H        128
#define G        512      // 4*H gate columns
#define BTILE    16
#define NTHREADS 512
#define NT_PER_WAVE 2

// ---- LDS layout (byte offsets into dynamic shared) ----
#define OFF_WHH1F 0        // w_hh1 f16 fragments: 32 ntg * 4 kf * 32 lanes * 16 halves = 131072 B
#define OFF_H0F   131072   // h0 state, A-frag layout, f16: 4*32*16*2 = 4096 B
#define OFF_H1F   135168   // h1 state, A-frag layout, f16: 4096 B
#define OFF_C0    139264   // c0 state f32 16x128 = 8192 B
#define OFF_C1    147456   // c1 state f32 16x128 = 8192 B
#define OFF_GB    155648   // gate buffer f32 16x512 = 32768 B
#define OFF_XC    188416   // input tile f32 16x2, double-buffered (2 x 128 B)
#define OFF_HR    188672   // decoder h row-major f16 16x128 = 4096 B
#define OFF_PAR   192768   // params tile f32 16x3 = 192 B
#define OFF_OW    192960   // out_w f32 2x128 = 1024 B
#define SMEM_BYTES 193984

__device__ __forceinline__ float sigm(float x) {
    return __builtin_amdgcn_rcpf(1.0f + __expf(-x));          // v_exp_f32 + v_rcp_f32
}
__device__ __forceinline__ float tanh_f(float x) {
    float r = __builtin_amdgcn_rcpf(__expf(2.0f * x) + 1.0f); // tanh = 1 - 2/(e^2x + 1)
    return fmaf(-2.0f, r, 1.0f);
}

// ---- Tensor Data Mover: DMA a 16x2 f32 tile (row stride T*2) from global -> LDS ----
// D# built per CDNA5 ISA ch.8: group0 {count=1, lds_addr, global_addr[56:0], type=2},
// group1 {data_size=4B, tensor_dim0=2, tensor_dim1=16, tile 2x16, dim0_stride=400}.
__device__ __forceinline__ void tdm_load_x_tile(const float* gptr, unsigned lds_off) {
    unsigned long long ga = (unsigned long long)(const void*)gptr;
    v4u g0;
    g0[0] = 1u;                                                 // count=1, user descriptor
    g0[1] = lds_off;                                            // lds_addr (bytes)
    g0[2] = (unsigned)(ga & 0xFFFFFFFFu);                       // global_addr[31:0]
    g0[3] = (unsigned)((ga >> 32) & 0x1FFFFFFu) | (2u << 30);   // global_addr[56:32] | type=2
    v8i g1;
    g1[0] = 0x00020000;        // workgroup_mask=0, data_size=2 (4 bytes)
    g1[1] = (int)(2u << 16);   // atomic_barrier_addr=0, tensor_dim0 = 2 (lo16)
    g1[2] = (int)(16u << 16);  // tensor_dim0 hi=0, tensor_dim1 = 16 (lo16)
    g1[3] = (int)(2u << 16);   // tensor_dim1 hi=0, tile_dim0 = 2
    g1[4] = 16;                // tile_dim1 = 16, tile_dim2 = 0
    g1[5] = 400;               // tensor_dim0_stride = T_ENC*2 (48-bit, low dword)
    g1[6] = 0;                 // stride hi, tensor_dim1_stride lo
    g1[7] = 0;
    v4i z4 = {0, 0, 0, 0};
#if defined(__clang_major__) && (__clang_major__ >= 23)
    v8i z8 = {0, 0, 0, 0, 0, 0, 0, 0};
    __builtin_amdgcn_tensor_load_to_lds(g0, g1, z4, z4, z8, 0);
#else
    __builtin_amdgcn_tensor_load_to_lds(g0, g1, z4, z4, 0);
#endif
}

// B-matrix fragment load: lane covers column n, K-range [kb, kb+16) contiguous in W row n.
__device__ __forceinline__ v16h load_bfrag(const float* p) {
    v16h r;
#pragma unroll
    for (int e = 0; e < 16; ++e) r[e] = (_Float16)p[e];
    return r;
}

// A-fragment element scatter: element (m, j) of 16x128 f16 matrix -> frag layout.
// A 16-bit 16x32 layout: lanes 0-15 hold K in {0..15}, lanes 16-31 hold K in {16..31} (contiguous, matches B).
__device__ __forceinline__ void store_h_frag(_Float16* hF, int m, int j, float v) {
    int kf = j >> 5, kl = j & 31;
    int lane = m + 16 * (kl >> 4);
    int e = kl & 15;
    hF[(kf * 32 + lane) * 16 + e] = (_Float16)v;
}

__device__ __forceinline__ float load_h_frag(const _Float16* hF, int m, int j) {
    int kf = j >> 5, kl = j & 31;
    int lane = m + 16 * (kl >> 4);
    int e = kl & 15;
    return (float)hF[(kf * 32 + lane) * 16 + e];
}

// LSTM pointwise: gates (i,f,g,o) in GB columns [0,128)[128,256)[256,384)[384,512)
__device__ __forceinline__ void lstm_pointwise(const float* GB, float* C, _Float16* HF,
                                               _Float16* HR, int tid) {
#pragma unroll
    for (int u = 0; u < 4; ++u) {
        int el = tid + NTHREADS * u;        // 2048 elements = 16 rows x 128
        int m = el >> 7, j = el & 127;
        float gi = GB[m * G + j];
        float gf = GB[m * G + 128 + j];
        float gg = GB[m * G + 256 + j];
        float go = GB[m * G + 384 + j];
        float c  = sigm(gf) * C[m * H + j] + sigm(gi) * tanh_f(gg);
        C[m * H + j] = c;
        float h = sigm(go) * tanh_f(c);
        store_h_frag(HF, m, j, h);
        if (HR) HR[m * H + j] = (_Float16)h;
    }
}

__global__ void __launch_bounds__(NTHREADS)
traj_lstm_kernel(const float* __restrict__ x_hist, const float* __restrict__ params,
                 const float* __restrict__ w_ih0, const float* __restrict__ w_hh0,
                 const float* __restrict__ b_ih0, const float* __restrict__ b_hh0,
                 const float* __restrict__ w_ih1, const float* __restrict__ w_hh1,
                 const float* __restrict__ b_ih1, const float* __restrict__ b_hh1,
                 const float* __restrict__ dec_w_ih, const float* __restrict__ dec_w_hh,
                 const float* __restrict__ dec_b_ih, const float* __restrict__ dec_b_hh,
                 const float* __restrict__ proj_w, const float* __restrict__ proj_b,
                 const float* __restrict__ out_w, const float* __restrict__ out_b,
                 float* __restrict__ out) {
    extern __shared__ char smem[];
    _Float16* WHH1F = (_Float16*)(smem + OFF_WHH1F);
    _Float16* H0F   = (_Float16*)(smem + OFF_H0F);
    _Float16* H1F   = (_Float16*)(smem + OFF_H1F);
    float*    C0    = (float*)(smem + OFF_C0);
    float*    C1    = (float*)(smem + OFF_C1);
    float*    GB    = (float*)(smem + OFF_GB);
    _Float16* HR    = (_Float16*)(smem + OFF_HR);
    float*    PAR   = (float*)(smem + OFF_PAR);
    float*    OW    = (float*)(smem + OFF_OW);

    const int tid  = threadIdx.x;
    const int wv   = tid >> 5;       // 0..15
    const int lane = tid & 31;
    const int li   = lane & 15;      // N within tile / column sub-index
    const int hi   = lane >> 4;      // lane-half selects K sub-range / M+8
    const int b0   = blockIdx.x * BTILE;
    const float* xbase = x_hist + (long)b0 * T_ENC * 2;

    // ---- stage w_hh1 f16 fragments into LDS (once) ----
    for (int fi = tid; fi < 32 * 4 * 32; fi += NTHREADS) {
        int ntg = fi >> 7, kf = (fi >> 5) & 3, L = fi & 31;
        int n = ntg * 16 + (L & 15);
        int kb = kf * 32 + 16 * (L >> 4);
        const float* p = w_hh1 + n * H + kb;
        _Float16* q = WHH1F + fi * 16;
#pragma unroll
        for (int e = 0; e < 16; ++e) q[e] = (_Float16)p[e];
    }
    // ---- zero states ----
    for (int i = tid; i < 2048; i += NTHREADS) {
        H0F[i] = (_Float16)0.f;  H1F[i] = (_Float16)0.f;
        C0[i] = 0.f;             C1[i] = 0.f;
    }

    // ---- per-wave register B-fragments + per-lane scalars ----
    v16h bW0[NT_PER_WAVE][4];   // w_hh0 (encoder), re-staged as dec_w_hh (decoder)
    v16h bWi1[NT_PER_WAVE][4];  // w_ih1
    float wx0[NT_PER_WAVE][2], bias0[NT_PER_WAVE], bias1[NT_PER_WAVE];
#pragma unroll
    for (int nt = 0; nt < NT_PER_WAVE; ++nt) {
        int n = wv * 32 + nt * 16 + li;
#pragma unroll
        for (int kf = 0; kf < 4; ++kf) {
            int kb = kf * 32 + 16 * hi;
            bW0[nt][kf]  = load_bfrag(w_hh0 + n * H + kb);
            bWi1[nt][kf] = load_bfrag(w_ih1 + n * H + kb);
        }
        wx0[nt][0] = w_ih0[n * 2 + 0];
        wx0[nt][1] = w_ih0[n * 2 + 1];
        bias0[nt] = b_ih0[n] + b_hh0[n];
        bias1[nt] = b_ih1[n] + b_hh1[n];
    }

    // Kick off TDM DMA of the t=0 input tile into XC buffer 0.
    if (wv == 0) tdm_load_x_tile(xbase, OFF_XC);
    __syncthreads();

    // ================= ENCODER: fused layer0 + layer1, 200 steps =================
    for (int t = 0; t < T_ENC; ++t) {
        if (wv == 0) {
            __builtin_amdgcn_s_wait_tensorcnt(0);        // x_t tile landed in LDS
            if (t + 1 < T_ENC)                           // prefetch next tile via TDM
                tdm_load_x_tile(xbase + (t + 1) * 2, OFF_XC + ((t + 1) & 1) * 128);
        }
        __syncthreads();
        const float* XCt = (const float*)(smem + OFF_XC + (t & 1) * 128);

        // ---- layer0 gates: bias + x@w_ih0^T (K=2 scalar) + h0@w_hh0^T (WMMA) ----
#pragma unroll
        for (int nt = 0; nt < NT_PER_WAVE; ++nt) {
            int n = wv * 32 + nt * 16 + li;
            v8f acc;
#pragma unroll
            for (int r = 0; r < 8; ++r) {
                int m = r + 8 * hi;
                acc[r] = bias0[nt] + XCt[m * 2] * wx0[nt][0] + XCt[m * 2 + 1] * wx0[nt][1];
            }
#pragma unroll
            for (int kf = 0; kf < 4; ++kf) {
                v16h a = *(const v16h*)&H0F[(kf * 32 + lane) * 16];
                acc = __builtin_amdgcn_wmma_f32_16x16x32_f16(false, a, false, bW0[nt][kf],
                                                             (short)0, acc, false, false);
            }
#pragma unroll
            for (int r = 0; r < 8; ++r) GB[(r + 8 * hi) * G + n] = acc[r];
        }
        __syncthreads();
        lstm_pointwise(GB, C0, H0F, nullptr, tid);   // -> new h0 (A-frag), c0
        __syncthreads();

        // ---- layer1 gates: bias + h0_new@w_ih1^T + h1@w_hh1^T (both WMMA) ----
#pragma unroll
        for (int nt = 0; nt < NT_PER_WAVE; ++nt) {
            int n = wv * 32 + nt * 16 + li;
            v8f acc;
#pragma unroll
            for (int r = 0; r < 8; ++r) acc[r] = bias1[nt];
#pragma unroll
            for (int kf = 0; kf < 4; ++kf) {
                v16h a = *(const v16h*)&H0F[(kf * 32 + lane) * 16];
                acc = __builtin_amdgcn_wmma_f32_16x16x32_f16(false, a, false, bWi1[nt][kf],
                                                             (short)0, acc, false, false);
            }
#pragma unroll
            for (int kf = 0; kf < 4; ++kf) {
                v16h a = *(const v16h*)&H1F[(kf * 32 + lane) * 16];
                v16h b = *(const v16h*)&WHH1F[((wv * 2 + nt) * 128 + kf * 32 + lane) * 16];
                acc = __builtin_amdgcn_wmma_f32_16x16x32_f16(false, a, false, b,
                                                             (short)0, acc, false, false);
            }
#pragma unroll
            for (int r = 0; r < 8; ++r) GB[(r + 8 * hi) * G + n] = acc[r];
        }
        __syncthreads();
        lstm_pointwise(GB, C1, H1F, nullptr, tid);   // -> h1, c1
        __syncthreads();
    }

    // ================= DECODER SETUP =================
    float* XC = (float*)(smem + OFF_XC);             // decoder uses buffer 0
    float dwx[NT_PER_WAVE][5], biasd[NT_PER_WAVE];
#pragma unroll
    for (int nt = 0; nt < NT_PER_WAVE; ++nt) {
        int n = wv * 32 + nt * 16 + li;
#pragma unroll
        for (int kf = 0; kf < 4; ++kf) {
            int kb = kf * 32 + 16 * hi;
            bW0[nt][kf] = load_bfrag(dec_w_hh + n * H + kb);   // reuse register space
        }
#pragma unroll
        for (int d = 0; d < 5; ++d) dwx[nt][d] = dec_w_ih[n * 5 + d];
        biasd[nt] = dec_b_ih[n] + dec_b_hh[n];
    }
    if (tid < 48) { int m = tid / 3, q = tid % 3; PAR[tid] = params[(long)(b0 + m) * 3 + q]; }
    if (tid < 256) OW[tid] = out_w[tid];
    if (tid < 32) {
        int m = tid >> 1, d = tid & 1;
        XC[m * 2 + d] = xbase[(long)m * T_ENC * 2 + (T_ENC - 1) * 2 + d];
    }
    __syncthreads();
    // h = h1 + params @ proj_w^T + proj_b ; c = c1 (in place). Decoder h lives in H0F + HR.
#pragma unroll
    for (int u = 0; u < 4; ++u) {
        int el = tid + NTHREADS * u;
        int m = el >> 7, j = el & 127;
        float hv = load_h_frag(H1F, m, j)
                 + PAR[m * 3 + 0] * proj_w[j * 3 + 0]
                 + PAR[m * 3 + 1] * proj_w[j * 3 + 1]
                 + PAR[m * 3 + 2] * proj_w[j * 3 + 2]
                 + proj_b[j];
        store_h_frag(H0F, m, j, hv);
        HR[m * H + j] = (_Float16)hv;
    }
    __syncthreads();

    // ================= DECODER: 200 steps =================
    for (int t = 0; t < T_DEC; ++t) {
#pragma unroll
        for (int nt = 0; nt < NT_PER_WAVE; ++nt) {
            int n = wv * 32 + nt * 16 + li;
            v8f acc;
#pragma unroll
            for (int r = 0; r < 8; ++r) {   // K=5 input contribution: [cur0,cur1,p0,p1,p2]
                int m = r + 8 * hi;
                acc[r] = biasd[nt]
                       + XC[m * 2 + 0] * dwx[nt][0] + XC[m * 2 + 1] * dwx[nt][1]
                       + PAR[m * 3 + 0] * dwx[nt][2] + PAR[m * 3 + 1] * dwx[nt][3]
                       + PAR[m * 3 + 2] * dwx[nt][4];
            }
#pragma unroll
            for (int kf = 0; kf < 4; ++kf) {
                v16h a = *(const v16h*)&H0F[(kf * 32 + lane) * 16];
                acc = __builtin_amdgcn_wmma_f32_16x16x32_f16(false, a, false, bW0[nt][kf],
                                                             (short)0, acc, false, false);
            }
#pragma unroll
            for (int r = 0; r < 8; ++r) GB[(r + 8 * hi) * G + n] = acc[r];
        }
        __syncthreads();
        lstm_pointwise(GB, C1, H0F, HR, tid);        // decoder h -> H0F + HR
        __syncthreads();
        if (wv == 0) {                                // cur = h @ out_w^T + out_b
            int m = li, n = hi;
            float s = out_b[n];
#pragma unroll 8
            for (int k = 0; k < H; ++k) s += (float)HR[m * H + k] * OW[n * H + k];
            out[(long)(b0 + m) * T_DEC * 2 + t * 2 + n] = s;
            XC[m * 2 + n] = s;
        }
        __syncthreads();
    }
}

extern "C" void kernel_launch(void* const* d_in, const int* in_sizes, int n_in,
                              void* d_out, int out_size, void* d_ws, size_t ws_size,
                              hipStream_t stream) {
    const float* x_hist   = (const float*)d_in[0];
    const float* params   = (const float*)d_in[1];
    // d_in[2] = pred_len (scalar, known 200)
    const float* w_ih0    = (const float*)d_in[3];
    const float* w_hh0    = (const float*)d_in[4];
    const float* b_ih0    = (const float*)d_in[5];
    const float* b_hh0    = (const float*)d_in[6];
    const float* w_ih1    = (const float*)d_in[7];
    const float* w_hh1    = (const float*)d_in[8];
    const float* b_ih1    = (const float*)d_in[9];
    const float* b_hh1    = (const float*)d_in[10];
    const float* dec_w_ih = (const float*)d_in[11];
    const float* dec_w_hh = (const float*)d_in[12];
    const float* dec_b_ih = (const float*)d_in[13];
    const float* dec_b_hh = (const float*)d_in[14];
    const float* proj_w   = (const float*)d_in[15];
    const float* proj_b   = (const float*)d_in[16];
    const float* out_w    = (const float*)d_in[17];
    const float* out_b    = (const float*)d_in[18];
    float* out = (float*)d_out;

    (void)hipFuncSetAttribute((const void*)traj_lstm_kernel,
                              hipFuncAttributeMaxDynamicSharedMemorySize, SMEM_BYTES);
    traj_lstm_kernel<<<dim3(B_TOT / BTILE), dim3(NTHREADS), SMEM_BYTES, stream>>>(
        x_hist, params, w_ih0, w_hh0, b_ih0, b_hh0, w_ih1, w_hh1, b_ih1, b_hh1,
        dec_w_ih, dec_w_hh, dec_b_ih, dec_b_hh, proj_w, proj_b, out_w, out_b, out);
}
